// ActorCriticSNN_60833916780544
// MI455X (gfx1250) — matrix operand order repaired and checked by reading, not compiled, not run on hardware
//
#include <hip/hip_runtime.h>

typedef __attribute__((ext_vector_type(16))) _Float16 v16h;
typedef __attribute__((ext_vector_type(8)))  _Float16 v8h;
typedef __attribute__((ext_vector_type(8)))  float    v8f;
typedef __attribute__((ext_vector_type(4)))  float    v4f;

#define T_STEPS 128
#define BATCH   4096
#define NIN     64
#define HID     32
#define AOUT    400
#define AOUTP   416          // actor K padded to 13 x 32 for WMMA chunks
#define WAVES   2
#define MTILE   16

// ---------------- LDS layout (bytes) ----------------
// Shared (per block): f16 transposed weights Wt[k][n], f32 biases, f16 action_mat.
#define OFF_WT_IN 0                          // 64x64 f16   = 8192
#define OFF_WT_L1 (OFF_WT_IN + 64*64*2)      // 64x32 f16   = 4096
#define OFF_WT_L2 (OFF_WT_L1 + 64*32*2)      // 6 x 32x32   = 12288
#define OFF_WT_A  (OFF_WT_L2 + 6*32*32*2)    // 32x400 f16  = 25600
#define OFF_WT_C  (OFF_WT_A  + 32*400*2)     // 32x16 f16   = 1024
#define OFF_B_IN  (OFF_WT_C  + 32*16*2)      // 64 f32
#define OFF_B_L   (OFF_B_IN  + 64*4)         // 7*32 f32
#define OFF_B_A   (OFF_B_L   + 7*32*4)       // 400 f32
#define OFF_AMH   (OFF_B_A   + 400*4)        // 416x16 f16  = 13312 (action_mat, N-padded)
#define OFF_WAVE  (OFF_AMH   + AOUTP*16*2)   // = 67264
// Per wave: spike tiles + membranes [n][m] f32.
#define PW_S      0                          // 16x64  f16 = 2048  (layer spikes)
#define PW_SA     (PW_S + 16*64*2)           // 16x416 f16 = 13312 (actor spikes, K-padded)
#define PW_MEMIN  (PW_SA + 16*AOUTP*2)       // 64x16 f32 = 4096
#define PW_MEMH   (PW_MEMIN + 64*16*4)       // 7 x 32x16 f32 = 14336
#define PW_MEMA   (PW_MEMH + 7*32*16*4)      // 400x16 f32 = 25600
#define PW_SIZE   (PW_MEMA + 400*16*4)       // 59392
#define SMEM_TOTAL (OFF_WAVE + WAVES*PW_SIZE) // 186048 B < 320 KB WGP LDS

__device__ static inline v8f vzero8() {
  v8f z;
#pragma unroll
  for (int i = 0; i < 8; ++i) z[i] = 0.0f;
  return z;
}

__device__ static inline v8f wmma_f16(v16h a, v16h b, v8f c) {
  // D = A(16x32 f16) x B(32x16 f16) + C(16x16 f32)
  return __builtin_amdgcn_wmma_f32_16x16x32_f16(false, a, false, b, (short)0, c,
                                                false, false);
}

// A fragment from a spike tile S[m][rs] f16 (row-major).
// 16-bit A 16x32 layout: lane m=l%16, kh=l/16; e<8 -> k=kbase+8kh+e,
// e>=8 -> k=kbase+16+8kh+(e-8). Each group is 8 contiguous halves -> ds_load_b128.
__device__ static inline v16h ldsA(const _Float16* S, int rs, int kbase, int lane) {
  const int m  = lane & 15;
  const int kh = lane >> 4;
  const v8h lo = *(const v8h*)(S + m * rs + kbase + 8 * kh);
  const v8h hi = *(const v8h*)(S + m * rs + kbase + 16 + 8 * kh);
  return __builtin_shufflevector(lo, hi, 0, 1, 2, 3, 4, 5, 6, 7, 8, 9, 10, 11,
                                 12, 13, 14, 15);
}

// B fragment from transposed weights Wt[k][npad] f16: lane n=l%16, kh=l/16;
// e<8 -> k=kc*32+8kh+e, e>=8 -> k=kc*32+16+8kh+(e-8).
__device__ static inline v16h ldsB(const _Float16* Wt, int npad, int kc, int n0,
                                   int lane) {
  const _Float16* p = Wt + (kc * 32 + 8 * (lane >> 4)) * npad + n0 + (lane & 15);
  v16h r;
#pragma unroll
  for (int e = 0; e < 8; ++e) {
    r[e]     = p[e * npad];
    r[8 + e] = p[(16 + e) * npad];
  }
  return r;
}

// LIF on one 16x16 D tile; membrane state in LDS, layout memT[n][16 m-floats].
// snntorch 'subtract' reset: reset from previous mem, spike from new mem.
__device__ static inline v8f lif_tile(float* memT, int n0, int lane, v8f cur,
                                      float bias, float beta, float thr) {
  const int n  = n0 + (lane & 15);
  const int mb = (lane >> 4) << 3;
  v4f* mp = (v4f*)(memT + n * 16 + mb);
  v4f ml = mp[0], mh = mp[1];
  float mv[8] = {ml[0], ml[1], ml[2], ml[3], mh[0], mh[1], mh[2], mh[3]};
  v8f spk;
#pragma unroll
  for (int v = 0; v < 8; ++v) {
    float mm  = mv[v];
    float rst = (mm > thr) ? thr : 0.0f;
    mm = fmaf(beta, mm, cur[v] + bias - rst);
    spk[v] = (mm > thr) ? 1.0f : 0.0f;
    mv[v]  = mm;
  }
  ml[0] = mv[0]; ml[1] = mv[1]; ml[2] = mv[2]; ml[3] = mv[3];
  mh[0] = mv[4]; mh[1] = mv[5]; mh[2] = mv[6]; mh[3] = mv[7];
  mp[0] = ml; mp[1] = mh;
  return spk;
}

// Scatter D-layout spikes (lane=n, vgpr=m) into S[m][rs] f16 for the next A load.
__device__ static inline void store_spikes(_Float16* S, int rs, int n0, int lane,
                                           v8f spk) {
  const int n  = n0 + (lane & 15);
  const int mb = (lane >> 4) << 3;
#pragma unroll
  for (int v = 0; v < 8; ++v) S[(mb + v) * rs + n] = (_Float16)spk[v];
}

__global__ __launch_bounds__(WAVES * 32) void snn_actor_critic(
    const float* __restrict__ inputs,
    const float* __restrict__ W_in,  const float* __restrict__ b_in,
    const float* __restrict__ W1, const float* __restrict__ b1,
    const float* __restrict__ W2, const float* __restrict__ b2,
    const float* __restrict__ W3, const float* __restrict__ b3,
    const float* __restrict__ W4, const float* __restrict__ b4,
    const float* __restrict__ W5, const float* __restrict__ b5,
    const float* __restrict__ W6, const float* __restrict__ b6,
    const float* __restrict__ W7, const float* __restrict__ b7,
    const float* __restrict__ W_actor, const float* __restrict__ b_actor,
    const float* __restrict__ W_critic, const float* __restrict__ b_critic,
    const float* __restrict__ beta_critic, const float* __restrict__ action_mat,
    float* __restrict__ out) {
  extern __shared__ char smem[];
  const int tid  = threadIdx.x;
  const int nth  = blockDim.x;
  const int lane = tid & 31;
  const int wave = tid >> 5;
  const int l16  = lane & 15;
  const int kh   = lane >> 4;
  const int mb   = kh << 3;

  _Float16* wtIn = (_Float16*)(smem + OFF_WT_IN);
  _Float16* wtL1 = (_Float16*)(smem + OFF_WT_L1);
  _Float16* wtA  = (_Float16*)(smem + OFF_WT_A);
  _Float16* wtC  = (_Float16*)(smem + OFF_WT_C);
  _Float16* amH  = (_Float16*)(smem + OFF_AMH);
  float* bInL = (float*)(smem + OFF_B_IN);
  float* bLL  = (float*)(smem + OFF_B_L);
  float* bAL  = (float*)(smem + OFF_B_A);

  // -------- stage weights (f32 -> f16, transposed Wt[k][n]) --------
  for (int i = tid; i < 64 * 64; i += nth) {
    int o = i >> 6, k = i & 63;
    wtIn[k * 64 + o] = (_Float16)W_in[i];
  }
  for (int i = tid; i < 32 * 64; i += nth) {
    int o = i >> 6, k = i & 63;
    wtL1[k * 32 + o] = (_Float16)W1[i];
  }
  {
    const float* WH[6] = {W2, W3, W4, W5, W6, W7};
    for (int l = 0; l < 6; ++l) {
      _Float16* wtl = (_Float16*)(smem + OFF_WT_L2 + l * (32 * 32 * 2));
      const float* W = WH[l];
      for (int i = tid; i < 32 * 32; i += nth) {
        int o = i >> 5, k = i & 31;
        wtl[k * 32 + o] = (_Float16)W[i];
      }
    }
  }
  for (int i = tid; i < AOUT * 32; i += nth) {
    int o = i >> 5, k = i & 31;
    wtA[k * AOUT + o] = (_Float16)W_actor[i];
  }
  for (int i = tid; i < 32 * 16; i += nth)  // critic padded to 16 cols, col0 live
    wtC[i] = ((i & 15) == 0) ? (_Float16)W_critic[i >> 4] : (_Float16)0.0f;
  // action_mat (400x4 f32) -> f16 B blocks amH[k][16], K padded to 416, N to 16
  for (int i = tid; i < AOUTP * 16; i += nth) {
    int k = i >> 4, a = i & 15;
    amH[i] = (k < AOUT && a < 4) ? (_Float16)action_mat[k * 4 + a]
                                 : (_Float16)0.0f;
  }
  for (int i = tid; i < 64; i += nth) bInL[i] = b_in[i];
  {
    const float* BH[7] = {b1, b2, b3, b4, b5, b6, b7};
    for (int i = tid; i < 7 * 32; i += nth) bLL[i] = BH[i >> 5][i & 31];
  }
  for (int i = tid; i < AOUT; i += nth) bAL[i] = b_actor[i];

  // per-wave state -> zero (covers actor spike pad columns + all membranes)
  {
    float* pw = (float*)(smem + OFF_WAVE + wave * PW_SIZE + PW_SA);
    const int nflt = (PW_SIZE - PW_SA) / 4;
    for (int i = lane; i < nflt; i += 32) pw[i] = 0.0f;
  }
  __syncthreads();

  _Float16* S     = (_Float16*)(smem + OFF_WAVE + wave * PW_SIZE + PW_S);
  _Float16* SA    = (_Float16*)(smem + OFF_WAVE + wave * PW_SIZE + PW_SA);
  float*    memIn = (float*)(smem + OFF_WAVE + wave * PW_SIZE + PW_MEMIN);
  float*    memH  = (float*)(smem + OFF_WAVE + wave * PW_SIZE + PW_MEMH);
  float*    memA  = (float*)(smem + OFF_WAVE + wave * PW_SIZE + PW_MEMA);

  // ---- hoist only the deep-chain invariants: layers 2..7 + critic B frags
  // (12+1 fragments = 104 VGPRs persistent; keeps peak pressure < 256 so the
  //  allocator stays out of s_set_vgpr_msb mode)
  v16h wbH[6][2];    // layers 2..7, [l][nt]
#pragma unroll
  for (int l = 0; l < 6; ++l) {
    const _Float16* wtl = (const _Float16*)(smem + OFF_WT_L2 + l * (32 * 32 * 2));
#pragma unroll
    for (int nt = 0; nt < 2; ++nt) wbH[l][nt] = ldsB(wtl, 32, 0, nt * 16, lane);
  }
  v16h wbC = ldsB(wtC, 16, 0, 0, lane);

  const int b0 = (blockIdx.x * WAVES + wave) * MTILE;
  const float betaC = beta_critic[0];
  const float biasC = b_critic[0];
  const float betaH[7] = {0.65f, 0.85f, 0.85f, 0.95f, 0.95f, 0.95f, 0.95f};
  const float thrH[7]  = {0.25f, 1.0f, 1.0f, 1.0f, 1.0f, 1.0f, 1.0f};

  float* valsOut = out;
  float* actOut  = out + (size_t)T_STEPS * BATCH;
  v8f mc = vzero8();

  for (int t = 0; t < T_STEPS; ++t) {
    const float* xr = inputs + ((size_t)t * BATCH + b0 + l16) * NIN;
    // x tile straight into A fragments (global b128 loads + cvt to f16)
    v16h ax[2];
#pragma unroll
    for (int kc = 0; kc < 2; ++kc) {
      const float* g0 = xr + kc * 32 + 8 * kh;
#pragma unroll
      for (int i = 0; i < 8; ++i) {
        ax[kc][i]     = (_Float16)g0[i];
        ax[kc][8 + i] = (_Float16)g0[16 + i];
      }
    }
    if (t + 1 < T_STEPS)  // gfx1250 global_prefetch_b8 of next step's tile
      __builtin_prefetch(xr + (size_t)BATCH * NIN, 0, 1);

    // ---- input layer: 16x64 = x(16x64) @ W_in^T, lif(0.65, 0.25) ----
#pragma unroll
    for (int nt = 0; nt < 4; ++nt) {
      v8f c = wmma_f16(ax[0], ldsB(wtIn, 64, 0, nt * 16, lane), vzero8());
      c     = wmma_f16(ax[1], ldsB(wtIn, 64, 1, nt * 16, lane), c);
      v8f spk = lif_tile(memIn, nt * 16, lane, c, bInL[nt * 16 + l16], 0.65f, 0.25f);
      store_spikes(S, 64, nt * 16, lane, spk);
    }

    // ---- layer 1: K=64 ----
    {
      v16h a0 = ldsA(S, 64, 0, lane);
      v16h a1 = ldsA(S, 64, 32, lane);
#pragma unroll
      for (int nt = 0; nt < 2; ++nt) {
        v8f c = wmma_f16(a0, ldsB(wtL1, 32, 0, nt * 16, lane), vzero8());
        c     = wmma_f16(a1, ldsB(wtL1, 32, 1, nt * 16, lane), c);
        v8f spk = lif_tile(memH, nt * 16, lane, c, bLL[nt * 16 + l16],
                           betaH[0], thrH[0]);
        store_spikes(S, 64, nt * 16, lane, spk);
      }
    }

    // ---- layers 2..7: K=32, weights resident in VGPRs ----
#pragma unroll
    for (int l = 1; l < 7; ++l) {
      v16h a = ldsA(S, 64, 0, lane);
#pragma unroll
      for (int nt = 0; nt < 2; ++nt) {
        v8f c = wmma_f16(a, wbH[l - 1][nt], vzero8());
        v8f spk = lif_tile(memH + l * (32 * 16), nt * 16, lane, c,
                           bLL[l * 32 + nt * 16 + l16], betaH[l], thrH[l]);
        store_spikes(S, 64, nt * 16, lane, spk);
      }
    }

    // ---- actor: 400 outputs = 25 N tiles, K=32; spikes scattered to SA ----
    v16h as = ldsA(S, 64, 0, lane);  // layer-7 spikes (also feeds critic)
#pragma unroll
    for (int nt = 0; nt < 25; ++nt) {
      v8f c = wmma_f16(as, ldsB(wtA, AOUT, 0, nt * 16, lane), vzero8());
      v8f spk = lif_tile(memA, nt * 16, lane, c, bAL[nt * 16 + l16], 0.95f, 1.0f);
      store_spikes(SA, AOUTP, nt * 16, lane, spk);
    }

    // ---- actions = spk_a(16x416) @ action_mat(416x16 padded): 13 WMMA chunks
    {
      v8f c = vzero8();
#pragma unroll
      for (int kc = 0; kc < AOUTP / 32; ++kc) {
        v16h a = ldsA(SA, AOUTP, kc * 32, lane);
        c = wmma_f16(a, ldsB(amH, 16, kc, 0, lane), c);
      }
      if (l16 < 4) {
#pragma unroll
        for (int v = 0; v < 8; ++v)
          actOut[((size_t)t * BATCH + b0 + mb + v) * 4 + l16] = c[v];
      }
    }

    // ---- critic: 1 output padded to one 16-wide tile; thr=100, membrane in VGPRs
    {
      v8f c = wmma_f16(as, wbC, vzero8());
#pragma unroll
      for (int v = 0; v < 8; ++v) {
        float mm  = mc[v];
        float rst = (mm > 100.0f) ? 100.0f : 0.0f;
        mm = fmaf(betaC, mm, c[v] + biasC - rst);
        mc[v] = mm;
      }
      if (l16 == 0) {
#pragma unroll
        for (int v = 0; v < 8; ++v)
          valsOut[(size_t)t * BATCH + b0 + mb + v] = mc[v];
      }
    }
  }
}

extern "C" void kernel_launch(void* const* d_in, const int* in_sizes, int n_in,
                              void* d_out, int out_size, void* d_ws, size_t ws_size,
                              hipStream_t stream) {
  (void)in_sizes; (void)n_in; (void)d_ws; (void)ws_size; (void)out_size;
  const float* inputs   = (const float*)d_in[0];
  const float* W_in     = (const float*)d_in[1];
  const float* b_in     = (const float*)d_in[2];
  const float* W1 = (const float*)d_in[3];   const float* b1 = (const float*)d_in[4];
  const float* W2 = (const float*)d_in[5];   const float* b2 = (const float*)d_in[6];
  const float* W3 = (const float*)d_in[7];   const float* b3 = (const float*)d_in[8];
  const float* W4 = (const float*)d_in[9];   const float* b4 = (const float*)d_in[10];
  const float* W5 = (const float*)d_in[11];  const float* b5 = (const float*)d_in[12];
  const float* W6 = (const float*)d_in[13];  const float* b6 = (const float*)d_in[14];
  const float* W7 = (const float*)d_in[15];  const float* b7 = (const float*)d_in[16];
  const float* W_actor  = (const float*)d_in[17];
  const float* b_actor  = (const float*)d_in[18];
  const float* W_critic = (const float*)d_in[19];
  const float* b_critic = (const float*)d_in[20];
  const float* beta_c   = (const float*)d_in[21];
  const float* act_mat  = (const float*)d_in[22];

  static bool attr_set = []() {
    hipFuncSetAttribute((const void*)snn_actor_critic,
                        hipFuncAttributeMaxDynamicSharedMemorySize, SMEM_TOTAL);
    return true;
  }();
  (void)attr_set;

  dim3 grid(BATCH / (WAVES * MTILE));  // 128 blocks x 2 waves x 16 batch rows
  dim3 block(WAVES * 32);
  snn_actor_critic<<<grid, block, SMEM_TOTAL, stream>>>(
      inputs, W_in, b_in, W1, b1, W2, b2, W3, b3, W4, b4, W5, b5, W6, b6, W7,
      b7, W_actor, b_actor, W_critic, b_critic, beta_c, act_mat, (float*)d_out);
}
